// WLN_Regressor_970662609320
// MI455X (gfx1250) — matrix operand order repaired and checked by reading, not compiled, not run on hardware
//
#include <hip/hip_runtime.h>

// ---------------------------------------------------------------------------
// WLN regressor for MI455X (gfx1250, wave32, WMMA f32<=f16 16x16x32).
//
// Roofline reasoning: after commuting gathers with the shared-weight matmuls,
// total GEMM work is ~60 GFLOP (f16 WMMA, f32 accum) and gather traffic is
// ~1.5 GB/depth served from the 192MB L2 (per-batch tables are 2.4-13 MB).
// So we want: (a) all dense math on v_wmma_f32_16x16x32_f16, (b) one
// lightweight gather/elementwise kernel per depth, (c) deterministic
// reductions (no float atomics).
// ---------------------------------------------------------------------------

typedef __attribute__((ext_vector_type(16))) _Float16 v16h;
typedef __attribute__((ext_vector_type(8)))  _Float16 v8h;
typedef __attribute__((ext_vector_type(8)))  float    v8f;

#define B_SZ   16
#define N_AT   2048
#define M_BD   4096
#define HID    300
#define HP     320     // padded hidden (10 K-chunks of 32, 20 N-tiles of 16)
#define AF     82
#define AFP    96      // padded atom-feature K
#define BF     6
#define BFP    32      // padded bond-feature K
#define MAXNB  10

// ---------------- WMMA fragment loads (per CDNA5 ISA 7.12.2 layouts) --------
// A (16x32 f16, MxK): lanes 0-15 hold row M=lane, K = kc+{0..7, 16..23};
//                     lanes 16-31 hold row M=lane-16, K = kc+{8..15, 24..31}.
__device__ __forceinline__ v16h load_a_frag(const _Float16* __restrict__ A,
                                            int lda, int row, int kc, int lane) {
  const int ko = (lane & 16) ? 8 : 0;
  const _Float16* p = A + (size_t)row * lda + kc + ko;
  v8h lo = *(const v8h*)(p);        // K = kc+ko .. kc+ko+7
  v8h hi = *(const v8h*)(p + 16);   // K = kc+ko+16 .. kc+ko+23
  return __builtin_shufflevector(lo, hi, 0,1,2,3,4,5,6,7,8,9,10,11,12,13,14,15);
}

// B (32x16 f16, KxN): lane holds column N=lane&15; lanes 0-15 K=kc+{0..7, 16..23},
// lanes 16-31 K=kc+{8..15, 24..31}.  Weights are stored transposed (N x Kpad) so this
// is two contiguous 16B loads per lane.
__device__ __forceinline__ v16h load_b_frag(const _Float16* __restrict__ WT,
                                            int ldk, int col, int kc, int lane) {
  const int ko = (lane & 16) ? 8 : 0;
  const _Float16* p = WT + (size_t)col * ldk + kc + ko;
  v8h lo = *(const v8h*)(p);
  v8h hi = *(const v8h*)(p + 16);
  return __builtin_shufflevector(lo, hi, 0,1,2,3,4,5,6,7,8,9,10,11,12,13,14,15);
}

// ---------------- GEMM:  Out = act( A@W1 [+ A2@W2] + bias ) -----------------
// Each wave computes a 16(M) x 64(N) strip: 1 A-frag reused across 4 WMMAs.
// 8 waves / block -> block covers 128 rows x 64 cols.  No divergence (EXEC
// all ones around WMMA).  Out is f16 with padded cols (>=300) exactly zero
// (padded weight cols are zero, bias is masked).
__global__ __launch_bounds__(256)
void gemm_f16_wmma(const _Float16* __restrict__ A,  int lda,  int K1,
                   const _Float16* __restrict__ WT1,
                   const _Float16* __restrict__ A2, int lda2, int K2,
                   const _Float16* __restrict__ WT2,
                   const float* __restrict__ bias, int do_relu,
                   _Float16* __restrict__ Out, int ldo)
{
  const int lane = threadIdx.x & 31;
  const int wave = threadIdx.x >> 5;
  const int row0 = (blockIdx.x * 8 + wave) * 16;
  const int nt0  = blockIdx.y * 4;
  const int r    = lane & 15;

  v8f acc[4] = {};

  for (int kc = 0; kc < K1; kc += 32) {
    v16h a = load_a_frag(A, lda, row0 + r, kc, lane);
#pragma unroll
    for (int t = 0; t < 4; ++t) {
      v16h b = load_b_frag(WT1, K1, (nt0 + t) * 16 + r, kc, lane);
      acc[t] = __builtin_amdgcn_wmma_f32_16x16x32_f16(
          false, a, false, b, (short)0, acc[t], false, false);
    }
  }
  for (int kc = 0; kc < K2; kc += 32) {     // optional second (A,W) pair
    v16h a = load_a_frag(A2, lda2, row0 + r, kc, lane);
#pragma unroll
    for (int t = 0; t < 4; ++t) {
      v16h b = load_b_frag(WT2, K2, (nt0 + t) * 16 + r, kc, lane);
      acc[t] = __builtin_amdgcn_wmma_f32_16x16x32_f16(
          false, a, false, b, (short)0, acc[t], false, false);
    }
  }

  // C/D layout: VGPR vg holds M = vg (lanes 0-15) or vg+8 (lanes 16-31),
  // column N = lane&15.
  const int moff = (lane & 16) ? 8 : 0;
#pragma unroll
  for (int t = 0; t < 4; ++t) {
    const int col = (nt0 + t) * 16 + r;
    const float bv = (bias != nullptr && col < HID) ? bias[col] : 0.f;
#pragma unroll
    for (int vg = 0; vg < 8; ++vg) {
      float v = acc[t][vg] + bv;
      if (do_relu) v = v > 0.f ? v : 0.f;
      Out[(size_t)(row0 + moff + vg) * ldo + col] = (_Float16)v;
    }
  }
}

// ---------------- Packing kernels ------------------------------------------
// Weight -> transposed padded f16:  dst[n*Kpad + k] = src[k*300 + n] (0-pad).
__global__ void pack_wT(_Float16* __restrict__ dst, const float* __restrict__ src,
                        int srcK, int Kpad)
{
  int idx = blockIdx.x * blockDim.x + threadIdx.x;
  if (idx >= HP * Kpad) return;
  int n = idx / Kpad, k = idx % Kpad;
  float v = (k < srcK && n < HID) ? src[(size_t)k * HID + n] : 0.f;
  dst[idx] = (_Float16)v;
}

// Activation f32 -> padded f16 rows.
__global__ void pack_a(_Float16* __restrict__ dst, const float* __restrict__ src,
                       int rows, int srcC, int Cpad)
{
  int idx = blockIdx.x * blockDim.x + threadIdx.x;
  if (idx >= rows * Cpad) return;
  int r = idx / Cpad, c = idx % Cpad;
  dst[idx] = (c < srcC) ? (_Float16)src[(size_t)r * srcC + c] : (_Float16)0.f;
}

// ---------------- Per-node gather kernels (L2-resident tables) -------------
// Depths 0,1:  nei_label = sum_j relu( A_u2[ag_j] + Pb_u2[bg_j] + b_u2 )
__global__ __launch_bounds__(HP)
void neighbor_label(const _Float16* __restrict__ Pu2,
                    const _Float16* __restrict__ Pbu2,
                    const float* __restrict__ b_u2,
                    const int* __restrict__ atom_graph,
                    const int* __restrict__ bond_graph,
                    const int* __restrict__ num_nbs,
                    _Float16* __restrict__ nei_out)
{
  const int node = blockIdx.x;              // 0 .. B*N-1
  const int b    = node >> 11;              // / 2048
  const int h    = threadIdx.x;             // 0 .. 319
  int nnb = num_nbs[node]; nnb = nnb < MAXNB ? nnb : MAXNB;
  const float bias = (h < HID) ? b_u2[h] : 0.f;
  float nlab = 0.f;
  for (int j = 0; j < nnb; ++j) {
    const int an = atom_graph[(size_t)node * MAXNB + j];
    const int bm = bond_graph[(size_t)node * MAXNB + j];
    const size_t ra = ((size_t)(b * N_AT + an)) * HP + h;
    const size_t rb = ((size_t)(b * M_BD + bm)) * HP + h;
    float p = (float)Pu2[ra] + (float)Pbu2[rb] + bias;
    nlab += p > 0.f ? p : 0.f;
  }
  nei_out[(size_t)node * HP + h] = (h < HID) ? (_Float16)nlab : (_Float16)0.f;
}

// Depth 2:  kernel = (sum_j A_na[ag_j]*Pb_nb[bg_j]) * A_sa * node_mask
__global__ __launch_bounds__(HP)
void neighbor_wlnkernel(const _Float16* __restrict__ Pna,
                        const _Float16* __restrict__ Pbnb,
                        const _Float16* __restrict__ Psa,
                        const float* __restrict__ node_mask,
                        const int* __restrict__ atom_graph,
                        const int* __restrict__ bond_graph,
                        const int* __restrict__ num_nbs,
                        float* __restrict__ kernelBuf)
{
  const int node = blockIdx.x;
  const int b    = node >> 11;
  const int h    = threadIdx.x;
  int nnb = num_nbs[node]; nnb = nnb < MAXNB ? nnb : MAXNB;
  float fnei = 0.f;
  for (int j = 0; j < nnb; ++j) {
    const int an = atom_graph[(size_t)node * MAXNB + j];
    const int bm = bond_graph[(size_t)node * MAXNB + j];
    fnei += (float)Pna[((size_t)(b * N_AT + an)) * HP + h] *
            (float)Pbnb[((size_t)(b * M_BD + bm)) * HP + h];
  }
  const float kv = fnei * (float)Psa[(size_t)node * HP + h] * node_mask[node];
  kernelBuf[(size_t)node * HP + h] = (h < HID) ? kv : 0.f;
}

// ---------------- Deterministic reductions ---------------------------------
__global__ __launch_bounds__(256)
void reduce_over_nodes(const float* __restrict__ kernelBuf, float* __restrict__ Ksum)
{
  const int b = blockIdx.x;    // 0..15
  const int h = blockIdx.y;    // 0..299
  __shared__ float sm[256];
  float acc = 0.f;
  for (int n = threadIdx.x; n < N_AT; n += 256)
    acc += kernelBuf[((size_t)(b * N_AT + n)) * HP + h];
  sm[threadIdx.x] = acc;
  __syncthreads();
  for (int s = 128; s > 0; s >>= 1) {
    if (threadIdx.x < s) sm[threadIdx.x] += sm[threadIdx.x + s];
    __syncthreads();
  }
  if (threadIdx.x == 0) Ksum[b * HID + h] = sm[0];
}

__global__ __launch_bounds__(256)
void final_proj(const float* __restrict__ Ksum, const float* __restrict__ W_out,
                const float* __restrict__ b_out, float* __restrict__ out)
{
  const int b = blockIdx.x;    // 0..15
  __shared__ float sm[256];
  float acc = 0.f;
  for (int h = threadIdx.x; h < HID; h += 256)
    acc += Ksum[b * HID + h] * W_out[h];
  sm[threadIdx.x] = acc;
  __syncthreads();
  for (int s = 128; s > 0; s >>= 1) {
    if (threadIdx.x < s) sm[threadIdx.x] += sm[threadIdx.x + s];
    __syncthreads();
  }
  if (threadIdx.x == 0) out[b] = sm[0] + b_out[0];
}

// ---------------------------------------------------------------------------
extern "C" void kernel_launch(void* const* d_in, const int* in_sizes, int n_in,
                              void* d_out, int out_size, void* d_ws, size_t ws_size,
                              hipStream_t stream)
{
  (void)in_sizes; (void)n_in; (void)out_size; (void)ws_size;

  const float* input_atom = (const float*)d_in[0];
  const float* input_bond = (const float*)d_in[1];
  const float* node_mask  = (const float*)d_in[2];
  const int*   atom_graph = (const int*)d_in[3];
  const int*   bond_graph = (const int*)d_in[4];
  const int*   num_nbs    = (const int*)d_in[5];
  const float* W_af  = (const float*)d_in[6];
  const float* W_na  = (const float*)d_in[7];
  const float* W_nb  = (const float*)d_in[8];
  const float* W_sa  = (const float*)d_in[9];
  const float* W_u2  = (const float*)d_in[10];
  const float* b_u2  = (const float*)d_in[11];
  const float* W_u1  = (const float*)d_in[12];
  const float* b_u1  = (const float*)d_in[13];
  const float* W_out = (const float*)d_in[14];
  const float* b_out = (const float*)d_in[15];
  float* out = (float*)d_out;

  // ---- workspace carve-out (~240 MB; every buffer fully overwritten) ----
  size_t off = 0;
  auto alloc = [&](size_t bytes) -> char* {
    char* p = (char*)d_ws + off;
    off += (bytes + 255) & ~(size_t)255;
    return p;
  };
  _Float16* WafT   = (_Float16*)alloc((size_t)HP * AFP * 2);
  _Float16* WnaT   = (_Float16*)alloc((size_t)HP * HP  * 2);
  _Float16* WsaT   = (_Float16*)alloc((size_t)HP * HP  * 2);
  _Float16* Wu2aT  = (_Float16*)alloc((size_t)HP * HP  * 2);
  _Float16* Wu2bT  = (_Float16*)alloc((size_t)HP * BFP * 2);
  _Float16* WnbT   = (_Float16*)alloc((size_t)HP * BFP * 2);
  _Float16* Wu1aT  = (_Float16*)alloc((size_t)HP * HP  * 2);
  _Float16* Wu1bT  = (_Float16*)alloc((size_t)HP * HP  * 2);
  _Float16* Aatom  = (_Float16*)alloc((size_t)B_SZ * N_AT * AFP * 2);
  _Float16* Abond  = (_Float16*)alloc((size_t)B_SZ * M_BD * BFP * 2);
  _Float16* atomA  = (_Float16*)alloc((size_t)B_SZ * N_AT * HP * 2);
  _Float16* atomB  = (_Float16*)alloc((size_t)B_SZ * N_AT * HP * 2);
  _Float16* P1     = (_Float16*)alloc((size_t)B_SZ * N_AT * HP * 2); // A_u2 / A_na
  _Float16* P2     = (_Float16*)alloc((size_t)B_SZ * N_AT * HP * 2); // A_sa
  _Float16* Pbnb   = (_Float16*)alloc((size_t)B_SZ * M_BD * HP * 2);
  _Float16* Pbu2   = (_Float16*)alloc((size_t)B_SZ * M_BD * HP * 2);
  _Float16* nei16  = (_Float16*)alloc((size_t)B_SZ * N_AT * HP * 2);
  float*    kBuf   = (float*)   alloc((size_t)B_SZ * N_AT * HP * 4);
  float*    Ksum   = (float*)   alloc((size_t)B_SZ * HID * 4);

  const int atomRows = B_SZ * N_AT;   // 32768 (2048 M-tiles)
  const int bondRows = B_SZ * M_BD;   // 65536 (4096 M-tiles)
  const dim3 gAtom(atomRows / 16 / 8, HP / 64);  // (256, 5)
  const dim3 gBond(bondRows / 16 / 8, HP / 64);  // (512, 5)

  // ---- pack weights (transposed, f16, zero-padded) ----
  auto wblk = [](int kpad) { return (HP * kpad + 255) / 256; };
  pack_wT<<<wblk(AFP), 256, 0, stream>>>(WafT,  W_af,            AF,  AFP);
  pack_wT<<<wblk(HP),  256, 0, stream>>>(WnaT,  W_na,            HID, HP);
  pack_wT<<<wblk(HP),  256, 0, stream>>>(WsaT,  W_sa,            HID, HP);
  pack_wT<<<wblk(HP),  256, 0, stream>>>(Wu2aT, W_u2,            HID, HP);
  pack_wT<<<wblk(BFP), 256, 0, stream>>>(Wu2bT, W_u2 + 300*HID,  BF,  BFP);
  pack_wT<<<wblk(BFP), 256, 0, stream>>>(WnbT,  W_nb,            BF,  BFP);
  pack_wT<<<wblk(HP),  256, 0, stream>>>(Wu1aT, W_u1,            HID, HP);
  pack_wT<<<wblk(HP),  256, 0, stream>>>(Wu1bT, W_u1 + 300*HID,  HID, HP);

  // ---- pack activations ----
  pack_a<<<(atomRows * AFP + 255) / 256, 256, 0, stream>>>(Aatom, input_atom, atomRows, AF, AFP);
  pack_a<<<(bondRows * BFP + 255) / 256, 256, 0, stream>>>(Abond, input_bond, bondRows, BF, BFP);

  // ---- atom_feat = relu(input_atom @ W_af) ----
  gemm_f16_wmma<<<gAtom, 256, 0, stream>>>(Aatom, AFP, AFP, WafT,
                                           nullptr, 0, 0, nullptr,
                                           nullptr, 1, atomA, HP);
  // ---- per-bond projections (depth-invariant) ----
  gemm_f16_wmma<<<gBond, 256, 0, stream>>>(Abond, BFP, BFP, WnbT,
                                           nullptr, 0, 0, nullptr,
                                           nullptr, 0, Pbnb, HP);
  gemm_f16_wmma<<<gBond, 256, 0, stream>>>(Abond, BFP, BFP, Wu2bT,
                                           nullptr, 0, 0, nullptr,
                                           nullptr, 0, Pbu2, HP);

  // ---- depths 0,1: only the label path feeds forward ----
  _Float16* cur = atomA;
  _Float16* nxt = atomB;
  for (int d = 0; d < 2; ++d) {
    gemm_f16_wmma<<<gAtom, 256, 0, stream>>>(cur, HP, HP, Wu2aT,
                                             nullptr, 0, 0, nullptr,
                                             nullptr, 0, P1, HP);
    neighbor_label<<<atomRows, HP, 0, stream>>>(P1, Pbu2, b_u2,
                                                atom_graph, bond_graph, num_nbs,
                                                nei16);
    gemm_f16_wmma<<<gAtom, 256, 0, stream>>>(cur, HP, HP, Wu1aT,
                                             nei16, HP, HP, Wu1bT,
                                             b_u1, 1, nxt, HP);
    _Float16* t = cur; cur = nxt; nxt = t;
  }

  // ---- depth 2: only the kernel path is used by the output ----
  gemm_f16_wmma<<<gAtom, 256, 0, stream>>>(cur, HP, HP, WnaT,
                                           nullptr, 0, 0, nullptr,
                                           nullptr, 0, P1, HP);
  gemm_f16_wmma<<<gAtom, 256, 0, stream>>>(cur, HP, HP, WsaT,
                                           nullptr, 0, 0, nullptr,
                                           nullptr, 0, P2, HP);
  neighbor_wlnkernel<<<atomRows, HP, 0, stream>>>(P1, Pbnb, P2, node_mask,
                                                  atom_graph, bond_graph, num_nbs,
                                                  kBuf);

  // ---- deterministic pooling + output projection ----
  reduce_over_nodes<<<dim3(B_SZ, HID), 256, 0, stream>>>(kBuf, Ksum);
  final_proj<<<B_SZ, 256, 0, stream>>>(Ksum, W_out, b_out, out);
}
